// Decoder_63024350101964
// MI455X (gfx1250) — matrix operand order repaired and checked by reading, not compiled, not run on hardware
//
#include <hip/hip_runtime.h>
#include <math.h>

// ---------------------------------------------------------------------------
// DA-RNN decoder for MI455X (gfx1250, wave32, WMMA).
// B=256, T-1=32, ENC=DEC=128, OUT=1.  Latency-bound sequential scan ->
// persistent kernel: 16 WGs x 8 waves.  Weights pre-swizzled into WMMA
// fragment order and held in REGISTERS for the whole scan; the per-WG
// step-invariant E1/encoder slices (128 KB each) are staged into LDS once
// (gfx1250 allows up to 320 KB LDS per workgroup), so the 32-step loop
// runs with essentially zero global-memory traffic.
// ---------------------------------------------------------------------------

typedef __attribute__((ext_vector_type(16))) __bf16 v16bf;
typedef __attribute__((ext_vector_type(8)))  __bf16 v8bf;
typedef __attribute__((ext_vector_type(8)))  float  v8f;

#define B_    256
#define T_    32
#define ENC_  128
#define DEC_  128
#define HC_   256   // 2*DEC
#define G4_   512   // 4*DEC

// workspace layout (bytes)
#define OFF_ENCBF   0u                        // [B][T][ENC] bf16        (2 MB)
#define OFF_E1      (2u*1024u*1024u)          // [B][T][ENC] bf16        (2 MB)
#define OFF_W1HCP   (4u*1024u*1024u)          // packed frags, 64 KB
#define OFF_W1ENCP  (OFF_W1HCP + 64u*1024u)   // packed frags, 32 KB
#define OFF_WHHP    (OFF_W1ENCP + 32u*1024u)  // packed frags, 128 KB
#define OFF_BG      (OFF_WHHP + 128u*1024u)   // [G4] f32                (2 KB)

// ---- fast transcendentals (CDNA5 VALU TRANS ops) --------------------------
#if __has_builtin(__builtin_amdgcn_tanhf)
static __device__ __forceinline__ float fast_tanh(float x) {
  return __builtin_amdgcn_tanhf(x);
}
#else
static __device__ __forceinline__ float fast_tanh(float x) { return tanhf(x); }
#endif

#if __has_builtin(__builtin_amdgcn_exp2f)
static __device__ __forceinline__ float fast_exp(float x) {
  return __builtin_amdgcn_exp2f(x * 1.44269504088896340736f);
}
#else
static __device__ __forceinline__ float fast_exp(float x) { return expf(x); }
#endif

#if __has_builtin(__builtin_amdgcn_rcpf)
static __device__ __forceinline__ float fast_rcp(float x) {
  return __builtin_amdgcn_rcpf(x);
}
#else
static __device__ __forceinline__ float fast_rcp(float x) { return 1.0f / x; }
#endif

static __device__ __forceinline__ float fast_sigmoid(float x) {
  return fast_rcp(1.0f + fast_exp(-x));
}

// ---- WMMA helpers ---------------------------------------------------------
static __device__ __forceinline__ v8f wmma_bf16_f32(v16bf a, v16bf b, v8f c) {
  return __builtin_amdgcn_wmma_f32_16x16x32_bf16(
      /*neg_a=*/false, a, /*neg_b=*/false, b,
      /*c_mod=*/(short)0, c, /*reuse_a=*/false, /*reuse_b=*/false);
}

// A fragment: 16x32 bf16 from row-major source (leading dim ld).
// Per-lane elements are one contiguous 16-elem run (matches packed B):
// K = k0+hi*16 + [0..15]  ->  2x 16-byte loads.
static __device__ __forceinline__ v16bf load_a_frag(const __bf16* base,
                                                    int lane, int k0, int ld) {
  const int m = lane & 15, hi = lane >> 4;
  const __bf16* p = base + (size_t)m * ld + k0 + hi * 16;
  const v8bf lo = *(const v8bf*)p;
  const v8bf hh = *(const v8bf*)(p + 8);
  return __builtin_shufflevector(lo, hh, 0, 1, 2, 3, 4, 5, 6, 7, 8, 9, 10, 11,
                                 12, 13, 14, 15);
}

// B fragment from pre-swizzled buffer: frag f -> 32 lanes x 16 contiguous bf16.
static __device__ __forceinline__ v16bf load_b_packed(const __bf16* base,
                                                      int lane, int f) {
  return *(const v16bf*)(base + ((size_t)f * 32 + lane) * 16);
}

// ---------------------------------------------------------------------------
// Kernel 1: bf16 conversion + pre-swizzled B-fragment weight buffers.
// Packed layout: element i -> j=i&15 (elem), lane=(i>>4)&31, f=i>>9 (=nb*NKB+kb)
// value = W[n=nb*16+(lane&15)][k = kb*32 + (lane>>4)*16 + j]
// ---------------------------------------------------------------------------
__global__ __launch_bounds__(256) void prep_convert(
    const float* __restrict__ enc, const float* __restrict__ W1,
    const float* __restrict__ Whh, const float* __restrict__ bih,
    const float* __restrict__ bhh, __bf16* __restrict__ encbf,
    __bf16* __restrict__ w1hcP, __bf16* __restrict__ w1encP,
    __bf16* __restrict__ whhP, float* __restrict__ bg) {
  const int g = blockIdx.x * blockDim.x + threadIdx.x;
  const int stride = gridDim.x * blockDim.x;
  for (int i = g; i < B_ * T_ * ENC_; i += stride) encbf[i] = (__bf16)enc[i];

  for (int i = g; i < HC_ * ENC_; i += stride) {  // W1hc: NKB=8, 8 nb
    const int j = i & 15, lane = (i >> 4) & 31, f = i >> 9;
    const int kb = f & 7, nb = f >> 3;
    const int n = nb * 16 + (lane & 15);
    const int k = kb * 32 + (lane >> 4) * 16 + j;
    w1hcP[i] = (__bf16)W1[n * (HC_ + ENC_) + k];
  }
  for (int i = g; i < ENC_ * ENC_; i += stride) {  // W1enc: NKB=4, 8 nb
    const int j = i & 15, lane = (i >> 4) & 31, f = i >> 9;
    const int kb = f & 3, nb = f >> 2;
    const int n = nb * 16 + (lane & 15);
    const int k = kb * 32 + (lane >> 4) * 16 + j;
    w1encP[i] = (__bf16)W1[n * (HC_ + ENC_) + HC_ + k];
  }
  for (int i = g; i < DEC_ * G4_; i += stride) {  // Whh: NKB=4, 32 nb
    const int j = i & 15, lane = (i >> 4) & 31, f = i >> 9;
    const int kb = f & 3, nb = f >> 2;
    const int n = nb * 16 + (lane & 15);
    const int k = kb * 32 + (lane >> 4) * 16 + j;
    whhP[i] = (__bf16)Whh[n * DEC_ + k];
  }
  for (int i = g; i < G4_; i += stride) bg[i] = bih[i] + bhh[i];
}

// ---------------------------------------------------------------------------
// Kernel 2: E1 = enc @ W1_enc^T + b1.  GEMM [8192,128]x[128,128].
// 64 blocks x 8 waves; each wave owns one 16-row M-tile, reuses its 4
// A-fragments across all 8 N-tiles.
// ---------------------------------------------------------------------------
__global__ __launch_bounds__(256) void prep_e1(
    const __bf16* __restrict__ encbf, const __bf16* __restrict__ w1encP,
    const float* __restrict__ b1, __bf16* __restrict__ E1) {
  const int tid = threadIdx.x, lane = tid & 31, wv = tid >> 5;
  const int mt = blockIdx.x * 8 + wv;  // 512 M-tiles total
  v16bf a[4];
#pragma unroll
  for (int kb = 0; kb < 4; ++kb)
    a[kb] = load_a_frag(encbf + (size_t)mt * 16 * ENC_, lane, kb * 32, ENC_);
#pragma unroll
  for (int nt = 0; nt < 8; ++nt) {
    v8f acc = {};
#pragma unroll
    for (int kb = 0; kb < 4; ++kb)
      acc = wmma_bf16_f32(a[kb], load_b_packed(w1encP, lane, nt * 4 + kb), acc);
    const int nl = lane & 15, hi = lane >> 4;
    const int n = nt * 16 + nl;
    const float bias = b1[n];
#pragma unroll
    for (int r = 0; r < 8; ++r) {
      const int row = mt * 16 + r + hi * 8;
      E1[(size_t)row * ENC_ + n] = (__bf16)(acc[r] + bias);
    }
  }
}

// ---------------------------------------------------------------------------
// Kernel 3: persistent 32-step decoder. One WG = 16 batches, 8 waves.
// Weight fragments in registers; E1/enc slices resident in LDS.
// ---------------------------------------------------------------------------
__global__ __launch_bounds__(256, 1) void decoder_main(
    const __bf16* __restrict__ encbf, const __bf16* __restrict__ E1,
    const __bf16* __restrict__ w1hcP, const __bf16* __restrict__ whhP,
    const float* __restrict__ bg, const float* __restrict__ wih,
    const float* __restrict__ w2g, const float* __restrict__ b2g,
    const float* __restrict__ fcW, const float* __restrict__ fcb,
    const float* __restrict__ fcfW, const float* __restrict__ fcfb,
    const float* __restrict__ yh, float* __restrict__ out) {
  // step-invariant per-WG activation slices, staged once (128 KB each)
  __shared__ __align__(16) __bf16 sE1[16 * T_ * ENC_];
  __shared__ __align__(16) __bf16 sEnc[16 * T_ * ENC_];
  // recurrent state + scratch (~57 KB)
  __shared__ __align__(32) __bf16 sHC[16][HC_];  // [h;c] bf16 (A src)  8 KB
  __shared__ float sC[16][DEC_];                 // f32 cell state      8 KB
  __shared__ float sCtx[16][ENC_];               // context             8 KB
  __shared__ float sBufA[16 * G4_];              // gates/attn alias   32 KB
  __shared__ float sW2[ENC_];
  __shared__ float sFcW[ENC_ + 1];
  __shared__ float sYt[16];

  float* hcw   = sBufA;              // [16][128]
  float* slog  = sBufA + 16 * ENC_;  // [16][32]
  float* sattn = slog + 16 * T_;     // [16][32]

  const int tid = threadIdx.x;
  const int lane = tid & 31, wv = tid >> 5;
  const int wg = blockIdx.x;
  const int m = tid >> 4, q = tid & 15;
  const int l16 = lane & 15;
  const int row2 = wv * 2 + (lane >> 4);  // 16 rows over 8 waves

  // ---- one-time stage of this WG's E1/enc slices into LDS (b128 copies) ---
  {
    const size_t base = (size_t)wg * 16 * T_ * ENC_;
    const uint4* gE = (const uint4*)(E1 + base);
    const uint4* gN = (const uint4*)(encbf + base);
    uint4* lE = (uint4*)sE1;
    uint4* lN = (uint4*)sEnc;
    for (int i = tid; i < 16 * T_ * ENC_ / 8; i += 256) {
      lE[i] = gE[i];
      lN[i] = gN[i];
    }
  }
  for (int i = tid; i < 16 * HC_; i += 256) (&sHC[0][0])[i] = (__bf16)0.0f;
  for (int i = tid; i < 16 * DEC_; i += 256) (&sC[0][0])[i] = 0.0f;
  if (tid < ENC_) sW2[tid] = w2g[tid];
  if (tid < ENC_ + 1) sFcW[tid] = fcW[tid];

  // ---- hoist all loop-invariant weight fragments & scalars into registers -
  v16bf bW1[8];  // phase-1: this wave's N-tile (nb=wv), K=256
#pragma unroll
  for (int kb = 0; kb < 8; ++kb)
    bW1[kb] = load_b_packed(w1hcP, lane, wv * 8 + kb);
  v16bf bWhh[16];  // phase-6: 4 N-tiles x 4 K-blocks
#pragma unroll
  for (int i6 = 0; i6 < 4; ++i6)
#pragma unroll
    for (int kb = 0; kb < 4; ++kb)
      bWhh[i6 * 4 + kb] = load_b_packed(whhP, lane, (wv * 4 + i6) * 4 + kb);
  float bgv[4], wihv[4];
#pragma unroll
  for (int i6 = 0; i6 < 4; ++i6) {
    const int n = (wv * 4 + i6) * 16 + l16;
    bgv[i6] = bg[n];
    wihv[i6] = wih[n];
  }
  const float b2 = b2g[0];
  const float fcb0 = fcb[0];

  __syncthreads();
  float hreg[8];

  for (int t = 0; t < T_; ++t) {
    // -- Phase 1: hcW = [h;c] @ W1hc^T ; wave wv owns N-tile wv (8 WMMA)
    {
      v8f acc = {};
#pragma unroll
      for (int kb = 0; kb < 8; ++kb)
        acc = wmma_bf16_f32(load_a_frag(&sHC[0][0], lane, kb * 32, HC_),
                            bW1[kb], acc);
      const int hi = lane >> 4;
#pragma unroll
      for (int r = 0; r < 8; ++r)
        hcw[(r + hi * 8) * ENC_ + wv * 16 + l16] = acc[r];
    }
    __syncthreads();

    // -- Phase 2: logits[b,tt] = w2 . tanh(E1 + hcW) + b2  (v_tanh_f32)
#pragma unroll
    for (int half = 0; half < 2; ++half) {
      const int tt = q + half * 16;
      const v8bf* ev = (const v8bf*)(sE1 + ((size_t)m * T_ + tt) * ENC_);
      const float* hw = hcw + m * ENC_;
      float acc = 0.0f;
#pragma unroll 2
      for (int c8 = 0; c8 < 16; ++c8) {
        const v8bf v = ev[c8];
#pragma unroll
        for (int j = 0; j < 8; ++j) {
          const int e = c8 * 8 + j;
          acc += sW2[e] * fast_tanh((float)v[j] + hw[e]);
        }
      }
      slog[m * T_ + tt] = acc + b2;
    }
    __syncthreads();

    // -- Phase 3: softmax over 32 positions; 2 rows/wave, shfl reductions
    {
      const float a0 = slog[row2 * T_ + l16];
      const float a1 = slog[row2 * T_ + l16 + 16];
      float mx = fmaxf(a0, a1);
#pragma unroll
      for (int d = 1; d < 16; d <<= 1) mx = fmaxf(mx, __shfl_xor(mx, d, 32));
      const float e0 = fast_exp(a0 - mx), e1v = fast_exp(a1 - mx);
      float s = e0 + e1v;
#pragma unroll
      for (int d = 1; d < 16; d <<= 1) s += __shfl_xor(s, d, 32);
      const float inv = fast_rcp(s);
      sattn[row2 * T_ + l16] = e0 * inv;
      sattn[row2 * T_ + l16 + 16] = e1v * inv;
    }
    __syncthreads();

    // -- Phase 4: ctx = attn-weighted sum of encoder states (LDS b128)
    {
      const int e0 = q * 8;
      float acc[8] = {};
      for (int tt = 0; tt < T_; ++tt) {
        const float w = sattn[m * T_ + tt];
        const v8bf ep =
            *(const v8bf*)(sEnc + ((size_t)m * T_ + tt) * ENC_ + e0);
#pragma unroll
        for (int j = 0; j < 8; ++j) acc[j] += w * (float)ep[j];
      }
#pragma unroll
      for (int j = 0; j < 8; ++j) sCtx[m][e0 + j] = acc[j];
    }
    __syncthreads();

    // -- Phase 5: y_tilde = fc([ctx, y_t]); 2 rows/wave, shfl reduction
    {
      float acc = 0.0f;
#pragma unroll
      for (int j = 0; j < 8; ++j) {
        const int e = l16 * 8 + j;
        acc += sFcW[e] * sCtx[row2][e];
      }
#pragma unroll
      for (int d = 1; d < 16; d <<= 1) acc += __shfl_xor(acc, d, 32);
      if (l16 == 0)
        sYt[row2] = acc + fcb0 +
                    sFcW[ENC_] * yh[(size_t)(wg * 16 + row2) * T_ + t];
    }
    __syncthreads();

    // -- Phase 6: gates = h @ Whh^T + y_tilde*W_ih + bg ; 4 N-tiles/wave
    {
#pragma unroll
      for (int i6 = 0; i6 < 4; ++i6) {
        v8f acc = {};
#pragma unroll
        for (int kb = 0; kb < 4; ++kb)
          acc = wmma_bf16_f32(load_a_frag(&sHC[0][0], lane, kb * 32, HC_),
                              bWhh[i6 * 4 + kb], acc);
        const int hi = lane >> 4;
        const int n = (wv * 4 + i6) * 16 + l16;
#pragma unroll
        for (int r = 0; r < 8; ++r) {
          const int mm = r + hi * 8;
          sBufA[mm * G4_ + n] = acc[r] + bgv[i6] + wihv[i6] * sYt[mm];
        }
      }
    }
    __syncthreads();

    // -- Phase 7: LSTM cell update (i,f,g,o)
    {
#pragma unroll
      for (int r = 0; r < 8; ++r) {
        const int d = q + r * 16;
        const float gi = sBufA[m * G4_ + d];
        const float gf = sBufA[m * G4_ + DEC_ + d];
        const float gg = sBufA[m * G4_ + 2 * DEC_ + d];
        const float go = sBufA[m * G4_ + 3 * DEC_ + d];
        const float c = fast_sigmoid(gf) * sC[m][d] +
                        fast_sigmoid(gi) * fast_tanh(gg);
        hreg[r] = fast_sigmoid(go) * fast_tanh(c);
        sC[m][d] = c;
      }
      __syncthreads();  // gate reads complete before sBufA reuse
#pragma unroll
      for (int r = 0; r < 8; ++r) {
        const int d = q + r * 16;
        sHC[m][d] = (__bf16)hreg[r];
        sHC[m][DEC_ + d] = (__bf16)sC[m][d];
        if (t == T_ - 1) sBufA[m * DEC_ + d] = hreg[r];  // stash f32 h
      }
    }
    __syncthreads();
  }

  // -- Final: out = fcf([h, ctx]) + fcf_b + y_history[:, -1]; shfl reduce
  {
    float acc = 0.0f;
#pragma unroll
    for (int j = 0; j < 8; ++j) {
      const int e = l16 * 8 + j;
      acc += fcfW[e] * sBufA[row2 * DEC_ + e];
      acc += fcfW[DEC_ + e] * sCtx[row2][e];
    }
#pragma unroll
    for (int d = 1; d < 16; d <<= 1) acc += __shfl_xor(acc, d, 32);
    if (l16 == 0) {
      const int bo = wg * 16 + row2;
      out[bo] = acc + fcfb[0] + yh[(size_t)bo * T_ + (T_ - 1)];
    }
  }
}

// ---------------------------------------------------------------------------
extern "C" void kernel_launch(void* const* d_in, const int* in_sizes, int n_in,
                              void* d_out, int out_size, void* d_ws,
                              size_t ws_size, hipStream_t stream) {
  const float* enc  = (const float*)d_in[0];
  const float* yh   = (const float*)d_in[1];
  const float* W1   = (const float*)d_in[2];
  const float* b1   = (const float*)d_in[3];
  const float* W2   = (const float*)d_in[4];
  const float* b2   = (const float*)d_in[5];
  const float* Wih  = (const float*)d_in[6];
  const float* Whh  = (const float*)d_in[7];
  const float* bih  = (const float*)d_in[8];
  const float* bhh  = (const float*)d_in[9];
  const float* fcW  = (const float*)d_in[10];
  const float* fcb  = (const float*)d_in[11];
  const float* fcfW = (const float*)d_in[12];
  const float* fcfb = (const float*)d_in[13];

  char* ws = (char*)d_ws;
  __bf16* encbf  = (__bf16*)(ws + OFF_ENCBF);
  __bf16* E1     = (__bf16*)(ws + OFF_E1);
  __bf16* w1hcP  = (__bf16*)(ws + OFF_W1HCP);
  __bf16* w1encP = (__bf16*)(ws + OFF_W1ENCP);
  __bf16* whhP   = (__bf16*)(ws + OFF_WHHP);
  float*  bgp    = (float*)(ws + OFF_BG);

  prep_convert<<<512, 256, 0, stream>>>(enc, W1, Whh, bih, bhh, encbf, w1hcP,
                                        w1encP, whhP, bgp);
  prep_e1<<<64, 256, 0, stream>>>(encbf, w1encP, b1, E1);
  decoder_main<<<16, 256, 0, stream>>>(encbf, E1, w1hcP, whhP, bgp, Wih, W2,
                                       b2, fcW, fcb, fcfW, fcfb, yh,
                                       (float*)d_out);
}
